// ProposalHead_14826227106357
// MI455X (gfx1250) — matrix operand order repaired and checked by reading, not compiled, not run on hardware
//
#include <hip/hip_runtime.h>

// ---------------------------------------------------------------------------
// PointNet++-style pipeline for gfx1250 (MI455X):
//   FPS -> ball query -> fused grouped MLP (bf16 WMMA, LDS-resident) x2
//   -> global SA (fused maxpool) -> linear heads.
// Matrix math: v_wmma_f32_16x16x32_bf16 (f32 accumulate).
// Weights for SA1/SA2 staged into LDS via global_load_async_to_lds_b128
// (ASYNCcnt), pipelined across layers with precise s_wait_asynccnt.
// ---------------------------------------------------------------------------

#define BB 32          // batch
#define NN 4096        // points

typedef __attribute__((ext_vector_type(16))) __bf16 v16bf;
typedef __attribute__((ext_vector_type(8)))  __bf16 v8bf;
typedef __attribute__((ext_vector_type(8)))  float  v8f;

__device__ __forceinline__ unsigned short f32_to_bf16(float f) {
  union { __bf16 h; unsigned short u; } cv;
  cv.h = (__bf16)f;          // native f32->bf16 (RNE) on gfx1250
  return cv.u;
}
__device__ __forceinline__ float bf16_to_f32(unsigned short h) {
  return __uint_as_float(((unsigned int)h) << 16);
}

// ---------------------------------------------------------------------------
// Weight prep: f32 [K x N] row-major -> bf16 [N x Kpad] (transposed, padded)
// ---------------------------------------------------------------------------
__global__ void prep_weight_kernel(const float* __restrict__ w, int K, int Nc,
                                   int Kpad, unsigned short* __restrict__ wt) {
  int e = blockIdx.x * blockDim.x + threadIdx.x;
  if (e >= Nc * Kpad) return;
  int n = e / Kpad, k = e % Kpad;
  float v = (k < K) ? w[k * Nc + n] : 0.0f;
  wt[e] = f32_to_bf16(v);
}

// ---------------------------------------------------------------------------
// Farthest point sampling: one block per batch, iterative argmax in LDS.
// ---------------------------------------------------------------------------
__global__ __launch_bounds__(256) void fps_kernel(const float* __restrict__ xyz,
                                                  int N, int npoint,
                                                  int* __restrict__ outIdx) {
  __shared__ float dist[NN];
  __shared__ float redV[256];
  __shared__ int   redI[256];
  __shared__ int   curIdx;
  const int b = blockIdx.x, tid = threadIdx.x;
  for (int n = tid; n < N; n += 256) dist[n] = 1e10f;
  if (tid == 0) curIdx = 0;
  __syncthreads();
  for (int i = 0; i < npoint; ++i) {
    const int cur = curIdx;
    if (tid == 0) outIdx[b * npoint + i] = cur;
    const float cx = xyz[(b * N + cur) * 3 + 0];
    const float cy = xyz[(b * N + cur) * 3 + 1];
    const float cz = xyz[(b * N + cur) * 3 + 2];
    float bestV = -1.0f; int bestI = 0;
    for (int n = tid; n < N; n += 256) {
      float dx = xyz[(b * N + n) * 3 + 0] - cx;
      float dy = xyz[(b * N + n) * 3 + 1] - cy;
      float dz = xyz[(b * N + n) * 3 + 2] - cz;
      float d  = dx * dx + dy * dy + dz * dz;
      float dn = fminf(dist[n], d);
      dist[n] = dn;
      if (dn > bestV) { bestV = dn; bestI = n; }
    }
    redV[tid] = bestV; redI[tid] = bestI;
    __syncthreads();
    for (int s = 128; s > 0; s >>= 1) {
      if (tid < s) {
        float v2 = redV[tid + s]; int i2 = redI[tid + s];
        if (v2 > redV[tid] || (v2 == redV[tid] && i2 < redI[tid])) {
          redV[tid] = v2; redI[tid] = i2;
        }
      }
      __syncthreads();
    }
    if (tid == 0) curIdx = redI[0];
    __syncthreads();
  }
}

// Gather selected centroids: nxyz[b][s] = xyz[b][cidx[b][s]]
__global__ void gather_xyz_kernel(const float* __restrict__ xyz,
                                  const int* __restrict__ cidx,
                                  int N, int S, int total,
                                  float* __restrict__ nxyz) {
  int e = blockIdx.x * blockDim.x + threadIdx.x;
  if (e >= total) return;
  int c = e % 3, s = (e / 3) % S, b = e / (3 * S);
  nxyz[e] = xyz[(b * N + cidx[b * S + s]) * 3 + c];
}

// ---------------------------------------------------------------------------
// Ball query: sequential scan keeps in-radius indices in ascending order,
// pads with first hit (same as sort-then-clip in the reference).
// ---------------------------------------------------------------------------
__global__ void ball_query_kernel(const float* __restrict__ xyz,
                                  const float* __restrict__ nxyz,
                                  int N, int S, int NSAMP, float r2, int total,
                                  int* __restrict__ gidx) {
  int e = blockIdx.x * blockDim.x + threadIdx.x;
  if (e >= total) return;
  int b = e / S;
  float cx = nxyz[e * 3 + 0], cy = nxyz[e * 3 + 1], cz = nxyz[e * 3 + 2];
  int* out = gidx + (size_t)e * NSAMP;
  int cnt = 0, firstIdx = 0;
  for (int n = 0; n < N && cnt < NSAMP; ++n) {
    float dx = xyz[(b * N + n) * 3 + 0] - cx;
    float dy = xyz[(b * N + n) * 3 + 1] - cy;
    float dz = xyz[(b * N + n) * 3 + 2] - cz;
    float d2 = dx * dx + dy * dy + dz * dz;
    if (d2 <= r2) { if (cnt == 0) firstIdx = n; out[cnt++] = n; }
  }
  for (; cnt < NSAMP; ++cnt) out[cnt] = firstIdx;
}

// ---------------------------------------------------------------------------
// Async DMA of a bf16 weight matrix [COUT x K] (row stride K in global) into
// LDS with row stride K+8 halves (pad kills bank-conflict periodicity).
// Each thread issues b128 async copies; per-wave instruction count is the
// uniform trip count => ASYNCcnt bookkeeping is exact.
// ---------------------------------------------------------------------------
template <int K, int COUT>
__device__ __forceinline__ void async_copy_weight(
    const unsigned short* __restrict__ src, unsigned short* dstLds) {
  constexpr int CH = K / 8;                      // b128 chunks per row
  static_assert((COUT * CH) % 256 == 0, "uniform trip count");
  const int tid = threadIdx.x;
  const unsigned base = (unsigned)(unsigned long long)dstLds;  // LDS byte off
  for (int e = tid; e < COUT * CH; e += 256) {
    const int row = e / CH, ch = e % CH;
    const unsigned lds_addr = base + (unsigned)(row * (K + 8) + ch * 8) * 2u;
    const unsigned short* g = src + (size_t)row * K + ch * 8;
    asm volatile("global_load_async_to_lds_b128 %0, %1, off"
                 :: "v"(lds_addr), "v"(g) : "memory");
  }
}

#define WAIT_ASYNCCNT(n) \
  asm volatile("s_wait_asynccnt %0" :: "n"(n) : "memory")

// ---------------------------------------------------------------------------
// One MLP layer on a 128-row LDS tile via WMMA bf16.
//   ldsIn : 128 x K bf16 (row stride K halves)
//   wt    : COUT x K bf16, row stride KB halves (LDS-staged or global)
//   ldsOut: 128 x COUT bf16, ReLU(acc + bias)
// 8 waves; wave w owns rows [16w, 16w+16).
// ---------------------------------------------------------------------------
template <int K, int KB, int COUT>
__device__ __forceinline__ void mlp_layer_wmma(const unsigned short* ldsIn,
                                               unsigned short* ldsOut,
                                               const unsigned short* __restrict__ wt,
                                               const float* __restrict__ bias) {
  static_assert(K % 32 == 0 && COUT % 16 == 0, "tile shapes");
  const int tid  = threadIdx.x;
  const int wave = tid >> 5;
  const int lane = tid & 31;
  const int hs   = lane >> 4;   // half-wave select
  const int ln   = lane & 15;
  const int m0   = wave * 16;
  const unsigned short* ap0 = ldsIn + (m0 + ln) * K + hs * 8;
#pragma unroll 1
  for (int nt = 0; nt < COUT / 16; ++nt) {
    const int col = nt * 16 + ln;
    const unsigned short* bp0 = wt + (size_t)col * KB + hs * 8;
    v8f acc = {};
#pragma unroll
    for (int kt = 0; kt < K / 32; ++kt) {
      v8bf alo = *(const v8bf*)(ap0 + kt * 32);
      v8bf ahi = *(const v8bf*)(ap0 + kt * 32 + 16);
      v16bf a = __builtin_shufflevector(alo, ahi, 0, 1, 2, 3, 4, 5, 6, 7, 8, 9,
                                        10, 11, 12, 13, 14, 15);
      v8bf blo = *(const v8bf*)(bp0 + kt * 32);
      v8bf bhi = *(const v8bf*)(bp0 + kt * 32 + 16);
      v16bf bm = __builtin_shufflevector(blo, bhi, 0, 1, 2, 3, 4, 5, 6, 7, 8, 9,
                                         10, 11, 12, 13, 14, 15);
      acc = __builtin_amdgcn_wmma_f32_16x16x32_bf16(false, a, false, bm,
                                                    (short)0, acc, false, false);
    }
    const float bv = bias[col];
#pragma unroll
    for (int i = 0; i < 8; ++i) {        // D: row = m0 + hs*8 + i, col
      const int row = m0 + hs * 8 + i;
      ldsOut[row * COUT + col] = f32_to_bf16(fmaxf(acc[i] + bv, 0.0f));
    }
  }
}

// ---------------------------------------------------------------------------
// Fused set-abstraction stage: async weight staging + gather -> 3 WMMA
// layers -> maxpool. One block = TILE_S centroids (TILE_S*NS == 128 rows).
// ---------------------------------------------------------------------------
template <int S, int NPTS_IN, int NS, int TILE_S, int CF, int K0, int C1,
          int C2, int C3>
__global__ __launch_bounds__(256) void sa_mlp_kernel(
    const float* __restrict__ xyz, const float* __restrict__ newxyz,
    const int* __restrict__ gidx, const float* __restrict__ feats,
    const unsigned short* __restrict__ w0, const float* __restrict__ b0,
    const unsigned short* __restrict__ w1, const float* __restrict__ b1,
    const unsigned short* __restrict__ w2, const float* __restrict__ b2,
    float* __restrict__ fout) {
  constexpr int ROWS = TILE_S * NS;
  static_assert(ROWS == 128, "8 waves x 16 rows");
  constexpr int CMAX = (C1 > C3) ? C1 : C3;
  constexpr int W0SZ = C1 * (K0 + 8);
  constexpr int W2SZ = C3 * (C2 + 8);
  constexpr int WBIG = (W0SZ > W2SZ) ? W0SZ : W2SZ;
  // per-wave async instruction counts per weight copy
  constexpr int N1 = (C2 * (C1 / 8)) / 256;
  constexpr int N2 = (C3 * (C2 / 8)) / 256;

  extern __shared__ char smemraw[];
  unsigned short* bufIn = (unsigned short*)smemraw;        // ROWS x K0
  unsigned short* bufA  = bufIn + ROWS * K0;               // ROWS x max(C1,C3)
  unsigned short* bufB  = bufA + ROWS * CMAX;              // ROWS x C2
  unsigned short* bufW0 = bufB + ROWS * C2;                // w0 then w2
  unsigned short* bufW1 = bufW0 + WBIG;                    // w1
  const int b   = blockIdx.y;
  const int s0  = blockIdx.x * TILE_S;
  const int tid = threadIdx.x;
  constexpr int CIN = 3 + CF;

  // Kick off DMA of layer-0/1 weights; overlap with the gather below.
  async_copy_weight<K0, C1>(w0, bufW0);
  async_copy_weight<C1, C2>(w1, bufW1);

  for (int e = tid; e < ROWS * K0; e += 256) bufIn[e] = 0;  // K padding
  __syncthreads();
  for (int e = tid; e < ROWS * CIN; e += 256) {
    const int r = e / CIN, c = e % CIN;
    const int t = r / NS, j = r % NS;
    const int s = s0 + t;
    const int idx = gidx[(b * S + s) * NS + j];
    float v;
    if (c < 3)
      v = xyz[(b * NPTS_IN + idx) * 3 + c] - newxyz[(b * S + s) * 3 + c];
    else
      v = feats[((size_t)b * NPTS_IN + idx) * CF + (c - 3)];
    bufIn[r * K0 + c] = f32_to_bf16(v);
  }
  WAIT_ASYNCCNT(N1);               // w0 landed (w1 may still be in flight)
  __syncthreads();
  mlp_layer_wmma<K0, K0 + 8, C1>(bufIn, bufA, bufW0, b0);
  __syncthreads();                 // all waves done reading bufW0
  async_copy_weight<C2, C3>(w2, bufW0);   // overlap w2 DMA with layer 1
  WAIT_ASYNCCNT(N2);               // w1 landed (only w2 outstanding)
  __syncthreads();
  mlp_layer_wmma<C1, C1 + 8, C2>(bufA, bufB, bufW1, b1);
  __syncthreads();
  WAIT_ASYNCCNT(0);                // w2 landed
  __syncthreads();
  mlp_layer_wmma<C2, C2 + 8, C3>(bufB, bufA, bufW0, b2);
  __syncthreads();
  // maxpool over the NS samples of each centroid (post-ReLU => >= 0)
  for (int e = tid; e < TILE_S * C3; e += 256) {
    const int t = e / C3, c = e % C3;
    const unsigned short* p = bufA + (t * NS) * C3 + c;
    float m = 0.0f;
    for (int j = 0; j < NS; ++j) m = fmaxf(m, bf16_to_f32(p[j * C3]));
    fout[((size_t)b * S + s0 + t) * C3 + c] = m;
  }
}

// ---------------------------------------------------------------------------
// Global SA stage: one block per batch (128 points). Weights too large for
// LDS -> direct global B-frags (prefetched); layer 2 fuses the max reduction.
// ---------------------------------------------------------------------------
template <int NP, int CINRAW, int K0, int C1, int C2, int C3>
__global__ __launch_bounds__(256) void sa3_kernel(
    const float* __restrict__ xyz2, const float* __restrict__ f2,
    const unsigned short* __restrict__ w0, const float* __restrict__ b0,
    const unsigned short* __restrict__ w1, const float* __restrict__ b1,
    const unsigned short* __restrict__ w2, const float* __restrict__ b2,
    float* __restrict__ feat) {
  static_assert(NP == 128, "one 128-row tile");
  extern __shared__ char smemraw[];
  unsigned short* bufIn = (unsigned short*)smemraw;   // NP x K0
  unsigned short* bufA  = bufIn + NP * K0;            // NP x C1
  unsigned short* bufB  = bufA + NP * C1;             // NP x C2
  float* scratch        = (float*)(bufB + NP * C2);   // 8 waves x 16 cols
  const int b = blockIdx.x;
  const int tid = threadIdx.x;
  const int wave = tid >> 5, lane = tid & 31;
  const int hs = lane >> 4, ln = lane & 15;
  const int m0 = wave * 16;
  constexpr int CF = CINRAW - 3;

  // Warm L2/WGP$ for the big layer-2 weight while we gather + compute.
  for (int e = tid; e < (C3 * C2) / 64; e += 256)
    __builtin_prefetch((const char*)(w2 + e * 64), 0, 1);

  for (int e = tid; e < NP * K0; e += 256) bufIn[e] = 0;
  __syncthreads();
  for (int e = tid; e < NP * CINRAW; e += 256) {
    const int r = e / CINRAW, c = e % CINRAW;
    float v = (c < 3) ? xyz2[(b * NP + r) * 3 + c]
                      : f2[((size_t)b * NP + r) * CF + (c - 3)];
    bufIn[r * K0 + c] = f32_to_bf16(v);
  }
  __syncthreads();
  mlp_layer_wmma<K0, K0, C1>(bufIn, bufA, w0, b0);
  __syncthreads();
  mlp_layer_wmma<C1, C1, C2>(bufA, bufB, w1, b1);
  __syncthreads();

  const unsigned short* ap0 = bufB + (m0 + ln) * C2 + hs * 8;
#pragma unroll 1
  for (int nt = 0; nt < C3 / 16; ++nt) {
    const int col = nt * 16 + ln;
    const unsigned short* bp0 = w2 + (size_t)col * C2 + hs * 8;
    v8f acc = {};
#pragma unroll
    for (int kt = 0; kt < C2 / 32; ++kt) {
      v8bf alo = *(const v8bf*)(ap0 + kt * 32);
      v8bf ahi = *(const v8bf*)(ap0 + kt * 32 + 16);
      v16bf a = __builtin_shufflevector(alo, ahi, 0, 1, 2, 3, 4, 5, 6, 7, 8, 9,
                                        10, 11, 12, 13, 14, 15);
      v8bf blo = *(const v8bf*)(bp0 + kt * 32);
      v8bf bhi = *(const v8bf*)(bp0 + kt * 32 + 16);
      v16bf bm = __builtin_shufflevector(blo, bhi, 0, 1, 2, 3, 4, 5, 6, 7, 8, 9,
                                         10, 11, 12, 13, 14, 15);
      acc = __builtin_amdgcn_wmma_f32_16x16x32_bf16(false, a, false, bm,
                                                    (short)0, acc, false, false);
    }
    const float bv = b2[col];
    float m = 0.0f;
#pragma unroll
    for (int i = 0; i < 8; ++i) m = fmaxf(m, fmaxf(acc[i] + bv, 0.0f));
    m = fmaxf(m, __shfl_xor(m, 16, 32));     // combine the two half-waves
    if (hs == 0) scratch[wave * 16 + ln] = m;
    __syncthreads();
    if (tid < 16) {
      float mm = scratch[tid];
#pragma unroll
      for (int w = 1; w < 8; ++w) mm = fmaxf(mm, scratch[w * 16 + tid]);
      feat[b * C3 + nt * 16 + tid] = mm;
    }
    __syncthreads();
  }
}

// ---------------------------------------------------------------------------
// Heads: cls (1024->1) and reg (1024->4); d_out = [cls(32) | reg(32x4)].
// ---------------------------------------------------------------------------
__global__ void head_kernel(const float* __restrict__ feat,
                            const float* __restrict__ cw,
                            const float* __restrict__ cb,
                            const float* __restrict__ rw,
                            const float* __restrict__ rb,
                            float* __restrict__ out) {
  int t = blockIdx.x * blockDim.x + threadIdx.x;
  if (t < BB) {
    float s = cb[0];
    for (int k = 0; k < 1024; ++k) s += feat[t * 1024 + k] * cw[k];
    out[t] = s;
  } else if (t < BB + BB * 4) {
    int e = t - BB, b = e / 4, o = e % 4;
    float s = rb[o];
    for (int k = 0; k < 1024; ++k) s += feat[b * 1024 + k] * rw[k * 4 + o];
    out[BB + e] = s;
  }
}

// ---------------------------------------------------------------------------
extern "C" void kernel_launch(void* const* d_in, const int* in_sizes, int n_in,
                              void* d_out, int out_size, void* d_ws,
                              size_t ws_size, hipStream_t stream) {
  (void)in_sizes; (void)n_in; (void)out_size; (void)ws_size;
  const float* xyz  = (const float*)d_in[0];
  const float* s1w0 = (const float*)d_in[1];  const float* s1b0 = (const float*)d_in[2];
  const float* s1w1 = (const float*)d_in[3];  const float* s1b1 = (const float*)d_in[4];
  const float* s1w2 = (const float*)d_in[5];  const float* s1b2 = (const float*)d_in[6];
  const float* s2w0 = (const float*)d_in[7];  const float* s2b0 = (const float*)d_in[8];
  const float* s2w1 = (const float*)d_in[9];  const float* s2b1 = (const float*)d_in[10];
  const float* s2w2 = (const float*)d_in[11]; const float* s2b2 = (const float*)d_in[12];
  const float* s3w0 = (const float*)d_in[13]; const float* s3b0 = (const float*)d_in[14];
  const float* s3w1 = (const float*)d_in[15]; const float* s3b1 = (const float*)d_in[16];
  const float* s3w2 = (const float*)d_in[17]; const float* s3b2 = (const float*)d_in[18];
  const float* clsw = (const float*)d_in[19]; const float* clsb = (const float*)d_in[20];
  const float* regw = (const float*)d_in[21]; const float* regb = (const float*)d_in[22];
  float* out = (float*)d_out;

  // ---- workspace carve (256B aligned) ----
  char* ws = (char*)d_ws;
  size_t off = 0;
  auto carve = [&](size_t bytes) -> void* {
    void* p = ws + off;
    off += (bytes + 255) & ~(size_t)255;
    return p;
  };
  int*   cidx1 = (int*)carve((size_t)BB * 512 * 4);
  float* nxyz1 = (float*)carve((size_t)BB * 512 * 3 * 4);
  int*   gidx1 = (int*)carve((size_t)BB * 512 * 32 * 4);
  float* f1    = (float*)carve((size_t)BB * 512 * 128 * 4);
  int*   cidx2 = (int*)carve((size_t)BB * 128 * 4);
  float* nxyz2 = (float*)carve((size_t)BB * 128 * 3 * 4);
  int*   gidx2 = (int*)carve((size_t)BB * 128 * 64 * 4);
  float* f2    = (float*)carve((size_t)BB * 128 * 256 * 4);
  float* feat  = (float*)carve((size_t)BB * 1024 * 4);
  unsigned short* tw1a = (unsigned short*)carve((size_t)64 * 32 * 2);
  unsigned short* tw1b = (unsigned short*)carve((size_t)64 * 64 * 2);
  unsigned short* tw1c = (unsigned short*)carve((size_t)128 * 64 * 2);
  unsigned short* tw2a = (unsigned short*)carve((size_t)128 * 160 * 2);
  unsigned short* tw2b = (unsigned short*)carve((size_t)128 * 128 * 2);
  unsigned short* tw2c = (unsigned short*)carve((size_t)256 * 128 * 2);
  unsigned short* tw3a = (unsigned short*)carve((size_t)256 * 288 * 2);
  unsigned short* tw3b = (unsigned short*)carve((size_t)512 * 256 * 2);
  unsigned short* tw3c = (unsigned short*)carve((size_t)1024 * 512 * 2);

  auto prep = [&](const float* w, int K, int Nc, int Kpad, unsigned short* dst) {
    int elems = Nc * Kpad;
    prep_weight_kernel<<<(elems + 255) / 256, 256, 0, stream>>>(w, K, Nc, Kpad, dst);
  };
  prep(s1w0, 3, 64, 32, tw1a);
  prep(s1w1, 64, 64, 64, tw1b);
  prep(s1w2, 64, 128, 64, tw1c);
  prep(s2w0, 131, 128, 160, tw2a);
  prep(s2w1, 128, 128, 128, tw2b);
  prep(s2w2, 128, 256, 128, tw2c);
  prep(s3w0, 259, 256, 288, tw3a);
  prep(s3w1, 256, 512, 256, tw3b);
  prep(s3w2, 512, 1024, 512, tw3c);

  // ---- SA1: N=4096 -> S=512, nsample=32, r=0.2 ----
  fps_kernel<<<BB, 256, 0, stream>>>(xyz, NN, 512, cidx1);
  {
    int total = BB * 512 * 3;
    gather_xyz_kernel<<<(total + 255) / 256, 256, 0, stream>>>(xyz, cidx1, NN, 512, total, nxyz1);
  }
  {
    int total = BB * 512;
    ball_query_kernel<<<(total + 255) / 256, 256, 0, stream>>>(xyz, nxyz1, NN, 512, 32, 0.04f, total, gidx1);
  }
  {
    // act: 128*(32+128+64); weights: max(64*40,128*72)=9216 + 64*72=4608
    constexpr size_t smem = ((size_t)128 * (32 + 128 + 64) + 9216 + 4608) * 2;  // ~85 KB
    sa_mlp_kernel<512, NN, 32, 4, 0, 32, 64, 64, 128>
        <<<dim3(512 / 4, BB), 256, smem, stream>>>(
            xyz, nxyz1, gidx1, (const float*)nullptr,
            tw1a, s1b0, tw1b, s1b1, tw1c, s1b2, f1);
  }

  // ---- SA2: N=512 -> S=128, nsample=64, r=0.4 ----
  fps_kernel<<<BB, 256, 0, stream>>>(nxyz1, 512, 128, cidx2);
  {
    int total = BB * 128 * 3;
    gather_xyz_kernel<<<(total + 255) / 256, 256, 0, stream>>>(nxyz1, cidx2, 512, 128, total, nxyz2);
  }
  {
    int total = BB * 128;
    ball_query_kernel<<<(total + 255) / 256, 256, 0, stream>>>(nxyz1, nxyz2, 512, 128, 64, 0.16f, total, gidx2);
  }
  {
    // act: 128*(160+256+128); weights: max(128*168,256*136)=34816 + 128*136=17408
    constexpr size_t smem = ((size_t)128 * (160 + 256 + 128) + 34816 + 17408) * 2;  // ~238 KB
    sa_mlp_kernel<128, 512, 64, 2, 128, 160, 128, 128, 256>
        <<<dim3(128 / 2, BB), 256, smem, stream>>>(
            nxyz1, nxyz2, gidx2, f1,
            tw2a, s2b0, tw2b, s2b1, tw2c, s2b2, f2);
  }

  // ---- SA3 (global): 128 points, [259->256->512->1024] + max ----
  {
    constexpr size_t smem =
        ((size_t)128 * 288 + 128 * 256 + 128 * 512) * 2 + 8 * 16 * 4;  // ~265 KB
    sa3_kernel<128, 259, 288, 256, 512, 1024><<<BB, 256, smem, stream>>>(
        nxyz2, f2, tw3a, s3b0, tw3b, s3b1, tw3c, s3b2, feat);
  }

  // ---- heads ----
  head_kernel<<<1, 192, 0, stream>>>(feat, clsw, clsb, regw, regb, out);
}
